// TransformerBlock_13597866459685
// MI455X (gfx1250) — matrix operand order repaired and checked by reading, not compiled, not run on hardware
//
#include <hip/hip_runtime.h>
#include <hip/hip_bf16.h>
#include <math.h>
#include <type_traits>

// ---------------------------------------------------------------------------
// Types
// ---------------------------------------------------------------------------
typedef __attribute__((ext_vector_type(8)))  float  v8f;
typedef __attribute__((ext_vector_type(8)))  float  f32x8;
typedef __attribute__((ext_vector_type(8)))  __bf16 bf16x8;
typedef __attribute__((ext_vector_type(16))) __bf16 bf16x16;
typedef __attribute__((ext_vector_type(4)))  int    v4i;

#define D_MODEL 1024
#define SEQ     2048
#define BATCH   4
#define NHEADS  16
#define DQKV    64
#define FFN     4096
#define MTOT    (BATCH * SEQ)        // 8192 tokens
#define QKV_N   (3 * NHEADS * DQKV)  // 3072

// LDS tile stride (bf16 elems): 32 K + 8 pad -> 80B rows (16B multiple)
#define LDS_LDK 40

// ---------------------------------------------------------------------------
// CDNA5 async global->LDS copy (ASYNCcnt path). Compile-safe fallback to a
// synchronous copy when the builtin is not declared by this toolchain.
// Probe result: arg0 is a non-const AS1 v4i* (global source), 4 args total.
// ---------------------------------------------------------------------------
#if defined(__has_builtin)
#if __has_builtin(__builtin_amdgcn_global_load_async_to_lds_b128)
#define HAS_ASYNC_LDS 1
#endif
#endif
#ifndef HAS_ASYNC_LDS
#define HAS_ASYNC_LDS 0
#endif

__device__ __forceinline__ void async_copy_b128(const __bf16* __restrict__ g,
                                                __bf16* l) {
#if HAS_ASYNC_LDS
  __builtin_amdgcn_global_load_async_to_lds_b128(
      (__attribute__((address_space(1))) v4i*)g,
      (__attribute__((address_space(3))) v4i*)l, 0, 0);
#else
  *(bf16x8*)l = *(const bf16x8*)g;
#endif
}

__device__ __forceinline__ void wait_async() {
#if HAS_ASYNC_LDS
#if __has_builtin(__builtin_amdgcn_s_wait_asynccnt)
  __builtin_amdgcn_s_wait_asynccnt(0);
#else
  asm volatile("s_wait_asynccnt 0x0" ::: "memory");
#endif
#endif
}

// ---------------------------------------------------------------------------
// WMMA helpers (CDNA5 16x16x32 bf16, fp32 accum)
// ---------------------------------------------------------------------------
__device__ __forceinline__ v8f wmma_bf16(bf16x16 a, bf16x16 b, v8f c) {
  return __builtin_amdgcn_wmma_f32_16x16x32_bf16(false, a, false, b, (short)0, c,
                                                 false, false);
}

// A-matrix 16x32 bf16 fragment (ISA 7.12.2):
// lanes 0-15 : M=lane,    K chunks [0..7]  and [16..23]
// lanes 16-31: M=lane-16, K chunks [8..15] and [24..31]
__device__ __forceinline__ bf16x16 ld_frag_a(const __bf16* base, int lda, int lane) {
  int r = lane & 15, half = lane >> 4;
  const __bf16* p = base + r * lda + half * 8;
  bf16x8 lo = *(const bf16x8*)(p);
  bf16x8 hi = *(const bf16x8*)(p + 16);
  return __builtin_shufflevector(lo, hi, 0, 1, 2, 3, 4, 5, 6, 7,
                                 8, 9, 10, 11, 12, 13, 14, 15);
}

// B-matrix 32x16 bf16 fragment sourced from a [N][K] (B-transposed) tile:
// lanes 0-15 : N=lane, K [0..7],[16..23] ; lanes 16-31: N=lane-16, K [8..15],[24..31]
__device__ __forceinline__ bf16x16 ld_frag_b(const __bf16* base, int lda, int lane) {
  int n = lane & 15, half = lane >> 4;
  const __bf16* p = base + n * lda + half * 8;
  bf16x8 lo = *(const bf16x8*)(p);
  bf16x8 hi = *(const bf16x8*)(p + 16);
  return __builtin_shufflevector(lo, hi, 0, 1, 2, 3, 4, 5, 6, 7,
                                 8, 9, 10, 11, 12, 13, 14, 15);
}

// Register-held 128x32 tile slice (16 elems/thread): thread t -> row t>>1,
// 16 contiguous K elems at (t&1)*16. Two 8-wide loads, two ds_store_b128.
template <typename T>
struct TileRegs {
  using VT = typename std::conditional<sizeof(T) == 4, f32x8, bf16x8>::type;
  VT v0, v1;
  __device__ __forceinline__ void load(const T* __restrict__ g, int ld, int t) {
    int r = t >> 1, c = (t & 1) * 16;
    const VT* p = (const VT*)(g + (size_t)r * ld + c);
    v0 = p[0];
    v1 = p[1];
  }
  __device__ __forceinline__ void store(__bf16 (*s)[LDS_LDK], int t) const {
    int r = t >> 1, c = (t & 1) * 16;
    bf16x8* d = (bf16x8*)&s[r][c];
    if constexpr (sizeof(T) == 4) {
      d[0] = __builtin_convertvector(v0, bf16x8);
      d[1] = __builtin_convertvector(v1, bf16x8);
    } else {
      d[0] = v0;
      d[1] = v1;
    }
  }
};

// Register-held 256x32 tile slice (32 elems/thread): thread t -> full row t.
template <typename T>
struct TileRegsA {
  using VT = typename std::conditional<sizeof(T) == 4, f32x8, bf16x8>::type;
  VT v[4];
  __device__ __forceinline__ void load(const T* __restrict__ g, int ld, int t) {
    const VT* p = (const VT*)(g + (size_t)t * ld);
#pragma unroll
    for (int i = 0; i < 4; ++i) v[i] = p[i];
  }
  __device__ __forceinline__ void store(__bf16 (*s)[LDS_LDK], int t) const {
    bf16x8* d = (bf16x8*)&s[t][0];
#pragma unroll
    for (int i = 0; i < 4; ++i) {
      if constexpr (sizeof(T) == 4)
        d[i] = __builtin_convertvector(v[i], bf16x8);
      else
        d[i] = v[i];
    }
  }
};

__device__ __forceinline__ float gelu_tanh(float g) {
  float g3 = g * g * g;
  return 0.5f * g * (1.0f + tanhf(0.7978845608028654f * (g + 0.044715f * g3)));
}

// ---------------------------------------------------------------------------
// LayerNorm: one workgroup per row of 1024 fp32
// ---------------------------------------------------------------------------
__global__ __launch_bounds__(256) void ln_kernel(const float* __restrict__ X,
                                                 const float* __restrict__ w,
                                                 float* __restrict__ out) {
  __shared__ float s_sum[8], s_sq[8];
  int row = blockIdx.x;
  int t = threadIdx.x, lane = t & 31, wave = t >> 5;
  const float4* xr = (const float4*)(X + (size_t)row * D_MODEL);
  float4 x = xr[t];
  float sum = x.x + x.y + x.z + x.w;
  float sq = x.x * x.x + x.y * x.y + x.z * x.z + x.w * x.w;
#pragma unroll
  for (int off = 16; off >= 1; off >>= 1) {
    sum += __shfl_xor(sum, off, 32);
    sq += __shfl_xor(sq, off, 32);
  }
  if (lane == 0) { s_sum[wave] = sum; s_sq[wave] = sq; }
  __syncthreads();
  float tot = 0.f, tq = 0.f;
#pragma unroll
  for (int i = 0; i < 8; ++i) { tot += s_sum[i]; tq += s_sq[i]; }
  float mu = tot * (1.0f / D_MODEL);
  float var = tq * (1.0f / D_MODEL) - mu * mu;
  float rs = rsqrtf(var + 1e-5f);
  float4 ww = ((const float4*)w)[t];
  float4 o;
  o.x = (x.x - mu) * rs * ww.x;
  o.y = (x.y - mu) * rs * ww.y;
  o.z = (x.z - mu) * rs * ww.z;
  o.w = (x.w - mu) * rs * ww.w;
  ((float4*)(out + (size_t)row * D_MODEL))[t] = o;
}

// ---------------------------------------------------------------------------
// Generic GEMM: C[M,N] = A[M,K] * W[N,K]^T (+ residual), via bf16 WMMA.
// Block = 256x128 C tile, 8 waves as 4(M)x2(N), wave tile 64x64, BK=32:
// 16 WMMA per 16 ds_load_b128 per k-step. Double-buffered LDS. bf16 A tiles
// stage via async global->LDS DMA (ASYNCcnt); fp32 tiles stage via registers
// with packed bf16 conversion. global_prefetch_b8 pulls the k+64 stage to L2.
// EPI: 0 = fp32 out, 1 = bf16 out, 2 = de-interleave QKV (Q/K row-major per
// (b,h); V transposed [bh][d][s] so attention P*V B-fragments are contiguous).
// ---------------------------------------------------------------------------
enum { EPI_F32 = 0, EPI_BF16 = 1, EPI_QKV = 2 };

template <typename AT, int EPI, bool HAS_RES>
__global__ __launch_bounds__(256) void gemm_bf16_kernel(
    const AT* __restrict__ A, const float* __restrict__ W,
    void* __restrict__ Cout, const float* __restrict__ Res,
    __bf16* __restrict__ Qb, __bf16* __restrict__ Kb, __bf16* __restrict__ Vb,
    int M, int N, int K) {
  constexpr bool A_BF16 = (sizeof(AT) == 2);
  __shared__ __bf16 As[2][256][LDS_LDK];
  __shared__ __bf16 Ws[2][128][LDS_LDK];
  int t = threadIdx.x, lane = t & 31, wave = t >> 5;
  int wm = wave >> 1, wn = wave & 1;
  int m0 = blockIdx.y * 256, n0 = blockIdx.x * 128;

  v8f acc[4][4];
#pragma unroll
  for (int i = 0; i < 4; ++i)
#pragma unroll
    for (int j = 0; j < 4; ++j) acc[i][j] = (v8f)0.0f;

  TileRegsA<AT> ta;
  TileRegs<float> tw;
  ta.load(A + (size_t)m0 * K, K, t);
  tw.load(W + (size_t)n0 * K, K, t);
  ta.store(As[0], t);
  tw.store(Ws[0], t);
  __syncthreads();

  int cur = 0;
  for (int k0 = 0; k0 < K; k0 += 32) {
    bool nxt = (k0 + 32) < K;
    if (nxt) {  // issue next tile's loads before compute
      if constexpr (A_BF16) {
        // bf16 A: async DMA straight into the other LDS buffer
        const __bf16* g = (const __bf16*)A + (size_t)(m0 + t) * K + k0 + 32;
        __bf16* l = &As[cur ^ 1][t][0];
#pragma unroll
        for (int i = 0; i < 4; ++i) async_copy_b128(g + 8 * i, l + 8 * i);
      } else {
        ta.load(A + (size_t)m0 * K + k0 + 32, K, t);
      }
      tw.load(W + (size_t)n0 * K + k0 + 32, K, t);
    }
    if (k0 + 64 < K) {  // pull the stage after that toward L2 (prefetch_b8)
      if constexpr (!A_BF16)
        __builtin_prefetch(A + ((size_t)m0 + t) * K + k0 + 64, 0, 1);
      __builtin_prefetch(W + ((size_t)n0 + (t >> 1)) * K + k0 + 64, 0, 1);
    }
    bf16x16 af[4], bfm[4];
#pragma unroll
    for (int i = 0; i < 4; ++i)
      af[i] = ld_frag_a(&As[cur][wm * 64 + i * 16][0], LDS_LDK, lane);
#pragma unroll
    for (int j = 0; j < 4; ++j)
      bfm[j] = ld_frag_b(&Ws[cur][wn * 64 + j * 16][0], LDS_LDK, lane);
#pragma unroll
    for (int i = 0; i < 4; ++i)
#pragma unroll
      for (int j = 0; j < 4; ++j) acc[i][j] = wmma_bf16(af[i], bfm[j], acc[i][j]);
    if (nxt) {
      if constexpr (!A_BF16) ta.store(As[cur ^ 1], t);
      tw.store(Ws[cur ^ 1], t);
    }
    if constexpr (A_BF16) wait_async();
    __syncthreads();
    cur ^= 1;
  }

  int half = lane >> 4, nl = lane & 15;
#pragma unroll
  for (int i = 0; i < 4; ++i)
#pragma unroll
    for (int j = 0; j < 4; ++j)
#pragma unroll
      for (int r = 0; r < 8; ++r) {
        int gr = m0 + wm * 64 + i * 16 + half * 8 + r;
        int gc = n0 + wn * 64 + j * 16 + nl;
        float v = acc[i][j][r];
        if constexpr (HAS_RES) v += Res[(size_t)gr * N + gc];
        if constexpr (EPI == EPI_F32) {
          ((float*)Cout)[(size_t)gr * N + gc] = v;
        } else if constexpr (EPI == EPI_BF16) {
          ((__bf16*)Cout)[(size_t)gr * N + gc] = (__bf16)v;
        } else {  // EPI_QKV: gc = h*192 + d*3 + sel
          int b = gr >> 11, s = gr & (SEQ - 1);
          int h = gc / 192;
          int rem = gc - h * 192;
          int d = rem / 3;
          int sel = rem - d * 3;
          int bh = b * NHEADS + h;
          __bf16 val = (__bf16)v;
          if (sel == 0)
            Qb[((size_t)bh * SEQ + s) * DQKV + d] = val;
          else if (sel == 1)
            Kb[((size_t)bh * SEQ + s) * DQKV + d] = val;
          else
            Vb[((size_t)bh * DQKV + d) * SEQ + s] = val;
        }
      }
}

// ---------------------------------------------------------------------------
// Flash-style attention. Grid: (SEQ/128, BATCH*NHEADS). 8 waves; each wave
// owns a 16-query tile; keys iterate in 64-chunks (16 WMMA per chunk) with
// online softmax. K/V chunks double-buffered via async global->LDS DMA.
// ---------------------------------------------------------------------------
__global__ __launch_bounds__(256) void attn_kernel(const __bf16* __restrict__ Qb,
                                                   const __bf16* __restrict__ Kb,
                                                   const __bf16* __restrict__ Vb,
                                                   __bf16* __restrict__ Aout) {
  __shared__ __bf16 Qs[128][72];     // [query][d], pre-scaled by 1/8
  __shared__ __bf16 Ks[2][64][72];   // [key][d]   (B^T layout for Q*K^T)
  __shared__ __bf16 Vs[2][64][72];   // [d][key]   (B^T layout for P*V)
  __shared__ __bf16 Ps[8][16][64];   // per-wave P tile (A layout source)

  int bh = blockIdx.y;
  int b = bh >> 4, h = bh & 15;
  int q0 = blockIdx.x * 128;
  int t = threadIdx.x, lane = t & 31, wave = t >> 5;
  int half = lane >> 4, nl = lane & 15;

  // Stage Q once, folding in the 1/sqrt(64) score scale.
  {
    int r = t >> 1, d0 = (t & 1) * 32;
    const bf16x8* src = (const bf16x8*)(Qb + ((size_t)bh * SEQ + q0 + r) * DQKV + d0);
#pragma unroll
    for (int i = 0; i < 4; ++i) {
      f32x8 f = __builtin_convertvector(src[i], f32x8) * 0.125f;
      *(bf16x8*)&Qs[r][d0 + 8 * i] = __builtin_convertvector(f, bf16x8);
    }
  }
  // K/V staging coords: thread t -> row t>>2, 16 contiguous elems at (t&3)*16
  int kr = t >> 2, kc = (t & 3) * 16;
  {
    const bf16x8* ks = (const bf16x8*)(Kb + ((size_t)bh * SEQ + kr) * DQKV + kc);
    *(bf16x8*)&Ks[0][kr][kc] = ks[0];
    *(bf16x8*)&Ks[0][kr][kc + 8] = ks[1];
    const bf16x8* vs = (const bf16x8*)(Vb + ((size_t)bh * DQKV + kr) * SEQ + kc);
    *(bf16x8*)&Vs[0][kr][kc] = vs[0];
    *(bf16x8*)&Vs[0][kr][kc + 8] = vs[1];
  }

  float m_prev[8], l_sum[8];
  v8f o[4];
#pragma unroll
  for (int r = 0; r < 8; ++r) { m_prev[r] = -1e30f; l_sum[r] = 0.0f; }
#pragma unroll
  for (int j = 0; j < 4; ++j) o[j] = (v8f)0.0f;
  __syncthreads();

  int cur = 0;
  for (int c0 = 0; c0 < SEQ; c0 += 64) {
    bool nxt = (c0 + 64) < SEQ;
    if (nxt) {  // async DMA next chunk into the other buffer
      const __bf16* kg = Kb + ((size_t)bh * SEQ + c0 + 64 + kr) * DQKV + kc;
      async_copy_b128(kg, &Ks[cur ^ 1][kr][kc]);
      async_copy_b128(kg + 8, &Ks[cur ^ 1][kr][kc + 8]);
      const __bf16* vg = Vb + ((size_t)bh * DQKV + kr) * SEQ + c0 + 64 + kc;
      async_copy_b128(vg, &Vs[cur ^ 1][kr][kc]);
      async_copy_b128(vg + 8, &Vs[cur ^ 1][kr][kc + 8]);
    }

    // S = Q * K^T : 16 queries x 64 keys (4 tiles), K-dim 64 (2 steps)
    v8f sc[4];
#pragma unroll
    for (int j = 0; j < 4; ++j) sc[j] = (v8f)0.0f;
#pragma unroll
    for (int s = 0; s < 2; ++s) {
      bf16x16 qa = ld_frag_a(&Qs[wave * 16][s * 32], 72, lane);
#pragma unroll
      for (int j = 0; j < 4; ++j) {
        bf16x16 kb_ = ld_frag_b(&Ks[cur][j * 16][s * 32], 72, lane);
        sc[j] = wmma_bf16(qa, kb_, sc[j]);
      }
    }

    // Online softmax (rows r + 8*half live across each 16-lane half-group)
#pragma unroll
    for (int r = 0; r < 8; ++r) {
      float v0 = sc[0][r], v1 = sc[1][r], v2 = sc[2][r], v3 = sc[3][r];
      float mx = fmaxf(fmaxf(v0, v1), fmaxf(v2, v3));
#pragma unroll
      for (int off = 1; off <= 8; off <<= 1) mx = fmaxf(mx, __shfl_xor(mx, off, 32));
      float mn = fmaxf(m_prev[r], mx);
      float p0 = __expf(v0 - mn), p1 = __expf(v1 - mn);
      float p2 = __expf(v2 - mn), p3 = __expf(v3 - mn);
      float sum = (p0 + p1) + (p2 + p3);
#pragma unroll
      for (int off = 1; off <= 8; off <<= 1) sum += __shfl_xor(sum, off, 32);
      float alpha = __expf(m_prev[r] - mn);
      l_sum[r] = l_sum[r] * alpha + sum;
      m_prev[r] = mn;
#pragma unroll
      for (int j = 0; j < 4; ++j) o[j][r] *= alpha;
      int pr = r + 8 * half;
      Ps[wave][pr][nl] = (__bf16)p0;
      Ps[wave][pr][16 + nl] = (__bf16)p1;
      Ps[wave][pr][32 + nl] = (__bf16)p2;
      Ps[wave][pr][48 + nl] = (__bf16)p3;
    }

    // O += P(16x64) * V(64x64); per-wave DS ops are in-order, no barrier
#pragma unroll
    for (int s = 0; s < 2; ++s) {
      bf16x16 pf = ld_frag_a(&Ps[wave][0][s * 32], 64, lane);
#pragma unroll
      for (int j = 0; j < 4; ++j) {
        bf16x16 vb_ = ld_frag_b(&Vs[cur][j * 16][s * 32], 72, lane);
        o[j] = wmma_bf16(pf, vb_, o[j]);
      }
    }

    wait_async();
    __syncthreads();
    cur ^= 1;
  }

  // Epilogue: normalize, store bf16 (consumed as A by the O-proj GEMM)
#pragma unroll
  for (int j = 0; j < 4; ++j)
#pragma unroll
    for (int r = 0; r < 8; ++r) {
      int g = q0 + wave * 16 + r + 8 * half;
      int gc = h * DQKV + j * 16 + nl;
      Aout[((size_t)b * SEQ + g) * D_MODEL + gc] = (__bf16)(o[j][r] / l_sum[r]);
    }
}

// ---------------------------------------------------------------------------
// Fused FC1 + GEGLU: h = (Xn*Wa^T) * gelu(Xn*Wg^T), Wa = W_fc1[0:4096],
// Wg = W_fc1[4096:8192]. Double-buffered. Output bf16 [8192][4096].
// ---------------------------------------------------------------------------
__global__ __launch_bounds__(256) void fc1_geglu_kernel(
    const float* __restrict__ Xn, const float* __restrict__ Wfc1,
    __bf16* __restrict__ H) {
  __shared__ __bf16 As[2][128][LDS_LDK];
  __shared__ __bf16 Wa[2][128][LDS_LDK];
  __shared__ __bf16 Wg[2][128][LDS_LDK];
  const int K = D_MODEL, N = FFN;
  int t = threadIdx.x, lane = t & 31, wave = t >> 5;
  int wm = wave >> 1, wn = wave & 1;
  int m0 = blockIdx.y * 128, n0 = blockIdx.x * 128;

  v8f ca[2][4], cg[2][4];
#pragma unroll
  for (int i = 0; i < 2; ++i)
#pragma unroll
    for (int j = 0; j < 4; ++j) { ca[i][j] = (v8f)0.0f; cg[i][j] = (v8f)0.0f; }

  TileRegs<float> tx, twa, twg;
  tx.load(Xn + (size_t)m0 * K, K, t);
  twa.load(Wfc1 + (size_t)n0 * K, K, t);
  twg.load(Wfc1 + (size_t)(FFN + n0) * K, K, t);
  tx.store(As[0], t);
  twa.store(Wa[0], t);
  twg.store(Wg[0], t);
  __syncthreads();

  int cur = 0;
  for (int k0 = 0; k0 < K; k0 += 32) {
    bool nxt = (k0 + 32) < K;
    if (nxt) {
      tx.load(Xn + (size_t)m0 * K + k0 + 32, K, t);
      twa.load(Wfc1 + (size_t)n0 * K + k0 + 32, K, t);
      twg.load(Wfc1 + (size_t)(FFN + n0) * K + k0 + 32, K, t);
    }
    if (k0 + 64 < K) {
      __builtin_prefetch(Xn + ((size_t)m0 + (t >> 1)) * K + k0 + 64, 0, 1);
      __builtin_prefetch(Wfc1 + ((size_t)n0 + (t >> 1)) * K + k0 + 64, 0, 1);
      __builtin_prefetch(Wfc1 + ((size_t)(FFN + n0) + (t >> 1)) * K + k0 + 64, 0, 1);
    }
    bf16x16 af[2];
    af[0] = ld_frag_a(&As[cur][wm * 32][0], LDS_LDK, lane);
    af[1] = ld_frag_a(&As[cur][wm * 32 + 16][0], LDS_LDK, lane);
#pragma unroll
    for (int j = 0; j < 4; ++j) {
      bf16x16 ba = ld_frag_b(&Wa[cur][wn * 64 + j * 16][0], LDS_LDK, lane);
      bf16x16 bg = ld_frag_b(&Wg[cur][wn * 64 + j * 16][0], LDS_LDK, lane);
#pragma unroll
      for (int i = 0; i < 2; ++i) {
        ca[i][j] = wmma_bf16(af[i], ba, ca[i][j]);
        cg[i][j] = wmma_bf16(af[i], bg, cg[i][j]);
      }
    }
    if (nxt) {
      tx.store(As[cur ^ 1], t);
      twa.store(Wa[cur ^ 1], t);
      twg.store(Wg[cur ^ 1], t);
    }
    __syncthreads();
    cur ^= 1;
  }

  int half = lane >> 4, nl = lane & 15;
#pragma unroll
  for (int i = 0; i < 2; ++i)
#pragma unroll
    for (int j = 0; j < 4; ++j)
#pragma unroll
      for (int r = 0; r < 8; ++r) {
        int gr = m0 + wm * 32 + i * 16 + half * 8 + r;
        int gc = n0 + wn * 64 + j * 16 + nl;
        float hv = ca[i][j][r] * gelu_tanh(cg[i][j][r]);
        H[(size_t)gr * N + gc] = (__bf16)hv;
      }
}

// ---------------------------------------------------------------------------
// Host-side orchestration
// ---------------------------------------------------------------------------
extern "C" void kernel_launch(void* const* d_in, const int* in_sizes, int n_in,
                              void* d_out, int out_size, void* d_ws, size_t ws_size,
                              hipStream_t stream) {
  (void)in_sizes; (void)n_in; (void)out_size; (void)ws_size;
  const float* X    = (const float*)d_in[0];
  const float* Wqkv = (const float*)d_in[1];
  const float* Wo   = (const float*)d_in[2];
  const float* Wfc1 = (const float*)d_in[3];
  const float* Wfc2 = (const float*)d_in[4];
  const float* ln1w = (const float*)d_in[5];
  const float* ln2w = (const float*)d_in[6];

  char* ws = (char*)d_ws;
  const size_t SZ_F32 = (size_t)MTOT * D_MODEL * sizeof(float);   // 32 MB
  const size_t SZ_HEAD = (size_t)MTOT * D_MODEL * sizeof(__bf16); // 16 MB
  float*  Xn    = (float*)(ws);
  __bf16* Qb    = (__bf16*)(ws + SZ_F32);
  __bf16* Kb    = (__bf16*)((char*)Qb + SZ_HEAD);
  __bf16* Vb    = (__bf16*)((char*)Kb + SZ_HEAD);
  __bf16* AttnO = (__bf16*)((char*)Vb + SZ_HEAD);
  float*  X1    = (float*)((char*)AttnO + SZ_HEAD);
  float*  Xn2   = (float*)((char*)X1 + SZ_F32);
  __bf16* Hbuf  = (__bf16*)((char*)Xn2 + SZ_F32);  // 64 MB

  // 1) LN1
  ln_kernel<<<MTOT, 256, 0, stream>>>(X, ln1w, Xn);
  // 2) QKV projection, de-interleaved bf16 epilogue
  gemm_bf16_kernel<float, EPI_QKV, false>
      <<<dim3(QKV_N / 128, MTOT / 256), 256, 0, stream>>>(
          Xn, Wqkv, nullptr, nullptr, Qb, Kb, Vb, MTOT, QKV_N, D_MODEL);
  // 3) Attention -> bf16
  attn_kernel<<<dim3(SEQ / 128, BATCH * NHEADS), 256, 0, stream>>>(Qb, Kb, Vb,
                                                                   AttnO);
  // 4) O-projection + residual -> X1 (fp32)
  gemm_bf16_kernel<__bf16, EPI_F32, true>
      <<<dim3(D_MODEL / 128, MTOT / 256), 256, 0, stream>>>(
          AttnO, Wo, (void*)X1, X, nullptr, nullptr, nullptr,
          MTOT, D_MODEL, D_MODEL);
  // 5) LN2
  ln_kernel<<<MTOT, 256, 0, stream>>>(X1, ln2w, Xn2);
  // 6) FC1 + GEGLU -> bf16 h
  fc1_geglu_kernel<<<dim3(FFN / 128, MTOT / 128), 256, 0, stream>>>(Xn2, Wfc1,
                                                                    Hbuf);
  // 7) FC2 + residual -> output (fp32)
  gemm_bf16_kernel<__bf16, EPI_F32, true>
      <<<dim3(D_MODEL / 128, MTOT / 256), 256, 0, stream>>>(
          Hbuf, Wfc2, d_out, X1, nullptr, nullptr, nullptr,
          MTOT, D_MODEL, FFN);
}
